// Processer_8916352107101
// MI455X (gfx1250) — matrix operand order repaired and checked by reading, not compiled, not run on hardware
//
#include <hip/hip_runtime.h>

// ---------------------------------------------------------------------------
// GNN message-passing (3 layers) on gfx1250 using bf16 WMMA (f32 accumulate).
// N=50000 nodes, E=800000 edges, D=H=128, L=3.
// ---------------------------------------------------------------------------

typedef __bf16 bf16_t;
typedef __attribute__((ext_vector_type(16))) __bf16 v16bf_t;
typedef __attribute__((ext_vector_type(8)))  __bf16 v8bf_t;
typedef __attribute__((ext_vector_type(4)))  __bf16 v4bf_t;
typedef __attribute__((ext_vector_type(8)))  float   v8f_t;

#define DD 128   // feature dim
#define BM 128   // rows per block tile
#define NLAYERS 3

static __device__ __forceinline__ v16bf_t cat8(v8bf_t a, v8bf_t b) {
  return __builtin_shufflevector(a, b, 0,1,2,3,4,5,6,7,8,9,10,11,12,13,14,15);
}

// ---------------------------------------------------------------------------
// One GEMM panel: this lane's 16-row A strip (LDS, contiguous K) times packed
// bf16 weights (global, fragment order). KT = K/32 k-chunks. acc = 8 N-tiles.
// A-frag layout (contiguous K, matching B pack): lanes0-15 K{0..15},
// lanes16-31 K{16..31} -> two b128 LDS loads per chunk.
// B packed so each lane reads 16 contiguous bf16 (two global b128 loads).
// ---------------------------------------------------------------------------
template<int KT>
static __device__ __forceinline__ void gemm_panel(const bf16_t* __restrict__ Arow,
                                                  const bf16_t* __restrict__ Bp,
                                                  int half, v8f_t acc[8]) {
#pragma unroll
  for (int kc = 0; kc < KT; ++kc) {
    const bf16_t* ap = Arow + kc * 32 + half * 16;
    v8bf_t a0 = *(const v8bf_t*)(ap);
    v8bf_t a1 = *(const v8bf_t*)(ap + 8);
    v16bf_t af = cat8(a0, a1);
    const bf16_t* bp = Bp + (size_t)kc * (8 * 32 * 16);
    if (kc + 1 < KT) __builtin_prefetch(Bp + (size_t)(kc + 1) * (8 * 32 * 16), 0, 1);
#pragma unroll
    for (int nt = 0; nt < 8; ++nt) {
      v16bf_t bfrag = *(const v16bf_t*)(bp + (size_t)nt * (32 * 16));
      acc[nt] = __builtin_amdgcn_wmma_f32_16x16x32_bf16(
          false, af, false, bfrag, (short)0, acc[nt], false, false);
    }
  }
}

// ---------------------------------------------------------------------------
// Weight prepack: fp32 [L][K][128] -> bf16 fragment order.
// dst[((l*KT + kt)*8 + nt)*32*16 + lane*16 + i] =
//      W[l][kt*32 + (lane>=16 ? 16 : 0) + i][nt*16 + (lane&15)]
// One 32-lane wave per 16x32x16 B tile; 8 waves per block.
// ---------------------------------------------------------------------------
__global__ void pack_w_kernel(const float* __restrict__ src, bf16_t* __restrict__ dst,
                              int K) {
  int warp = (blockIdx.x * blockDim.x + threadIdx.x) >> 5;
  int lane = threadIdx.x & 31;
  int tilesPerLayer = (K / 32) * 8;
  int total = NLAYERS * tilesPerLayer;
  if (warp >= total) return;
  int l  = warp / tilesPerLayer;
  int t  = warp % tilesPerLayer;
  int kt = t >> 3, nt = t & 7;
  int n  = nt * 16 + (lane & 15);
  int kb = kt * 32 + ((lane & 16) ? 16 : 0);
  const float* s = src + (size_t)l * K * DD;
  bf16_t* o = dst + ((size_t)warp * 32 + lane) * 16;
#pragma unroll
  for (int i = 0; i < 16; ++i) o[i] = (bf16_t)s[(size_t)(kb + i) * DD + n];
}

__global__ void zero_f32_kernel(float* __restrict__ p, long long n) {
  long long i = (long long)blockIdx.x * blockDim.x + threadIdx.x;
  if (i < n) p[i] = 0.0f;
}

// ---------------------------------------------------------------------------
// Edge MLP + LayerNorm + residual-ef + atomic scatter-add into agg.
// Block = 256 threads (8 wave32), tile = 128 edges.
// ---------------------------------------------------------------------------
__global__ __launch_bounds__(256)
void edge_mlp_kernel(const float* __restrict__ x,
                     float* __restrict__ ef,           // in/out (residual)
                     float* __restrict__ agg,          // atomic target, pre-zeroed
                     const int* __restrict__ ei,       // [2,E]
                     const bf16_t* __restrict__ pW0,
                     const bf16_t* __restrict__ pW1,
                     const bf16_t* __restrict__ pW2,
                     const float* __restrict__ b0,
                     const float* __restrict__ b1,
                     const float* __restrict__ b2,
                     const float* __restrict__ g,
                     const float* __restrict__ beta,
                     int layer, long long E) {
  constexpr int KA  = 3 * DD;          // 384
  constexpr int LDA = KA + 8;          // padded, keeps 16B row alignment
  constexpr int LDH = DD + 8;
  extern __shared__ __align__(16) char smem_raw[];
  bf16_t* Abuf = (bf16_t*)smem_raw;                 // [128][LDA]
  bf16_t* Hbuf = Abuf + BM * LDA;                   // [128][LDH]
  __shared__ int sIdx[BM];
  __shared__ int dIdx[BM];

  const int tid  = threadIdx.x;
  const int lane = tid & 31;
  const int w    = tid >> 5;
  const int half = lane >> 4;
  const int ln   = lane & 15;
  const long long ebase = (long long)blockIdx.x * BM;

  if (tid < BM) {
    long long e = ebase + tid;
    int s_ = 0, d_ = 0;
    if (e < E) { s_ = ei[e]; d_ = ei[E + e]; }
    sIdx[tid] = s_;
    dIdx[tid] = d_;
  }
  __syncthreads();

  // Gather [x[src] | x[dst] | ef] -> LDS bf16, float4-vectorized.
  for (int it = tid; it < BM * (KA / 4); it += 256) {
    int row = it / (KA / 4);
    int c4  = (it % (KA / 4)) * 4;
    long long erow = ebase + row;
    if (erow >= E) erow = E - 1;
    const float* sp;
    if (c4 < DD)            sp = x  + (size_t)sIdx[row] * DD + c4;
    else if (c4 < 2 * DD)   sp = x  + (size_t)dIdx[row] * DD + (c4 - DD);
    else                    sp = ef + (size_t)erow * DD + (c4 - 2 * DD);
    float4 v = *(const float4*)sp;
    v4bf_t o;
    o[0] = (bf16_t)v.x; o[1] = (bf16_t)v.y; o[2] = (bf16_t)v.z; o[3] = (bf16_t)v.w;
    *(v4bf_t*)(Abuf + (size_t)row * LDA + c4) = o;
  }
  __syncthreads();

  const bf16_t* Arow = Abuf + (size_t)(w * 16 + ln) * LDA;
  const bf16_t* Hrow = Hbuf + (size_t)(w * 16 + ln) * LDH;
  v8f_t acc[8];

  // GEMM1: [128 x 384] @ W0 -> relu -> Hbuf
#pragma unroll
  for (int nt = 0; nt < 8; ++nt) {
    float bv = b0[layer * DD + nt * 16 + ln];
#pragma unroll
    for (int r = 0; r < 8; ++r) acc[nt][r] = bv;
  }
  gemm_panel<KA / 32>(Arow, pW0 + ((size_t)layer * (KA / 32) * 8 * 32 + lane) * 16,
                      half, acc);
#pragma unroll
  for (int nt = 0; nt < 8; ++nt) {
    int n = nt * 16 + ln;
#pragma unroll
    for (int r = 0; r < 8; ++r) {
      float v = fmaxf(acc[nt][r], 0.0f);
      Hbuf[(size_t)(w * 16 + r + half * 8) * LDH + n] = (bf16_t)v;   // wave-local strip
    }
  }

  // GEMM2: [128 x 128] @ W1 -> relu -> Hbuf
#pragma unroll
  for (int nt = 0; nt < 8; ++nt) {
    float bv = b1[layer * DD + nt * 16 + ln];
#pragma unroll
    for (int r = 0; r < 8; ++r) acc[nt][r] = bv;
  }
  gemm_panel<DD / 32>(Hrow, pW1 + ((size_t)layer * (DD / 32) * 8 * 32 + lane) * 16,
                      half, acc);
#pragma unroll
  for (int nt = 0; nt < 8; ++nt) {
    int n = nt * 16 + ln;
#pragma unroll
    for (int r = 0; r < 8; ++r) {
      float v = fmaxf(acc[nt][r], 0.0f);
      Hbuf[(size_t)(w * 16 + r + half * 8) * LDH + n] = (bf16_t)v;
    }
  }

  // GEMM3: [128 x 128] @ W2 -> LayerNorm
#pragma unroll
  for (int nt = 0; nt < 8; ++nt) {
    float bv = b2[layer * DD + nt * 16 + ln];
#pragma unroll
    for (int r = 0; r < 8; ++r) acc[nt][r] = bv;
  }
  gemm_panel<DD / 32>(Hrow, pW2 + ((size_t)layer * (DD / 32) * 8 * 32 + lane) * 16,
                      half, acc);

  float mu[8], rs[8];
#pragma unroll
  for (int r = 0; r < 8; ++r) {
    float s = 0.0f, q = 0.0f;
#pragma unroll
    for (int nt = 0; nt < 8; ++nt) { float v = acc[nt][r]; s += v; q += v * v; }
#pragma unroll
    for (int m = 1; m < 16; m <<= 1) { s += __shfl_xor(s, m, 32); q += __shfl_xor(q, m, 32); }
    float mean = s * (1.0f / 128.0f);
    float var  = q * (1.0f / 128.0f) - mean * mean;
    mu[r] = mean;
    rs[r] = rsqrtf(var + 1e-5f);
  }

#pragma unroll
  for (int nt = 0; nt < 8; ++nt) {
    int n = nt * 16 + ln;
    float gv = g[layer * DD + n];
    float bv = beta[layer * DD + n];
#pragma unroll
    for (int r = 0; r < 8; ++r) {
      int m = w * 16 + r + half * 8;
      long long e = ebase + m;
      if (e < E) {
        float y = (acc[nt][r] - mu[r]) * rs[r] * gv + bv;
        size_t off = (size_t)e * DD + n;
        ef[off] += y;                                      // ef residual (row owned)
        atomicAdd(agg + (size_t)dIdx[m] * DD + n, y);      // scatter-add
      }
    }
  }
}

// ---------------------------------------------------------------------------
// Node MLP + LayerNorm + in-place residual on x.
// ---------------------------------------------------------------------------
__global__ __launch_bounds__(256)
void node_mlp_kernel(float* __restrict__ x,               // in/out
                     const float* __restrict__ agg,
                     const bf16_t* __restrict__ pW0,
                     const bf16_t* __restrict__ pW1,
                     const bf16_t* __restrict__ pW2,
                     const float* __restrict__ b0,
                     const float* __restrict__ b1,
                     const float* __restrict__ b2,
                     const float* __restrict__ g,
                     const float* __restrict__ beta,
                     int layer, long long N) {
  constexpr int KA  = 2 * DD;          // 256
  constexpr int LDA = KA + 8;
  constexpr int LDH = DD + 8;
  extern __shared__ __align__(16) char smem_raw[];
  bf16_t* Abuf = (bf16_t*)smem_raw;
  bf16_t* Hbuf = Abuf + BM * LDA;

  const int tid  = threadIdx.x;
  const int lane = tid & 31;
  const int w    = tid >> 5;
  const int half = lane >> 4;
  const int ln   = lane & 15;
  const long long nbase = (long long)blockIdx.x * BM;

  for (int it = tid; it < BM * (KA / 4); it += 256) {
    int row = it / (KA / 4);
    int c4  = (it % (KA / 4)) * 4;
    long long nrow = nbase + row;
    if (nrow >= N) nrow = N - 1;
    const float* sp = (c4 < DD) ? (x + (size_t)nrow * DD + c4)
                                : (agg + (size_t)nrow * DD + (c4 - DD));
    float4 v = *(const float4*)sp;
    v4bf_t o;
    o[0] = (bf16_t)v.x; o[1] = (bf16_t)v.y; o[2] = (bf16_t)v.z; o[3] = (bf16_t)v.w;
    *(v4bf_t*)(Abuf + (size_t)row * LDA + c4) = o;
  }
  __syncthreads();

  const bf16_t* Arow = Abuf + (size_t)(w * 16 + ln) * LDA;
  const bf16_t* Hrow = Hbuf + (size_t)(w * 16 + ln) * LDH;
  v8f_t acc[8];

#pragma unroll
  for (int nt = 0; nt < 8; ++nt) {
    float bv = b0[layer * DD + nt * 16 + ln];
#pragma unroll
    for (int r = 0; r < 8; ++r) acc[nt][r] = bv;
  }
  gemm_panel<KA / 32>(Arow, pW0 + ((size_t)layer * (KA / 32) * 8 * 32 + lane) * 16,
                      half, acc);
#pragma unroll
  for (int nt = 0; nt < 8; ++nt) {
    int n = nt * 16 + ln;
#pragma unroll
    for (int r = 0; r < 8; ++r) {
      float v = fmaxf(acc[nt][r], 0.0f);
      Hbuf[(size_t)(w * 16 + r + half * 8) * LDH + n] = (bf16_t)v;
    }
  }

#pragma unroll
  for (int nt = 0; nt < 8; ++nt) {
    float bv = b1[layer * DD + nt * 16 + ln];
#pragma unroll
    for (int r = 0; r < 8; ++r) acc[nt][r] = bv;
  }
  gemm_panel<DD / 32>(Hrow, pW1 + ((size_t)layer * (DD / 32) * 8 * 32 + lane) * 16,
                      half, acc);
#pragma unroll
  for (int nt = 0; nt < 8; ++nt) {
    int n = nt * 16 + ln;
#pragma unroll
    for (int r = 0; r < 8; ++r) {
      float v = fmaxf(acc[nt][r], 0.0f);
      Hbuf[(size_t)(w * 16 + r + half * 8) * LDH + n] = (bf16_t)v;
    }
  }

#pragma unroll
  for (int nt = 0; nt < 8; ++nt) {
    float bv = b2[layer * DD + nt * 16 + ln];
#pragma unroll
    for (int r = 0; r < 8; ++r) acc[nt][r] = bv;
  }
  gemm_panel<DD / 32>(Hrow, pW2 + ((size_t)layer * (DD / 32) * 8 * 32 + lane) * 16,
                      half, acc);

  float mu[8], rs[8];
#pragma unroll
  for (int r = 0; r < 8; ++r) {
    float s = 0.0f, q = 0.0f;
#pragma unroll
    for (int nt = 0; nt < 8; ++nt) { float v = acc[nt][r]; s += v; q += v * v; }
#pragma unroll
    for (int m = 1; m < 16; m <<= 1) { s += __shfl_xor(s, m, 32); q += __shfl_xor(q, m, 32); }
    float mean = s * (1.0f / 128.0f);
    float var  = q * (1.0f / 128.0f) - mean * mean;
    mu[r] = mean;
    rs[r] = rsqrtf(var + 1e-5f);
  }

#pragma unroll
  for (int nt = 0; nt < 8; ++nt) {
    int n = nt * 16 + ln;
    float gv = g[layer * DD + n];
    float bv = beta[layer * DD + n];
#pragma unroll
    for (int r = 0; r < 8; ++r) {
      int m = w * 16 + r + half * 8;
      long long nrow = nbase + m;
      if (nrow < N) {
        float y = (acc[nt][r] - mu[r]) * rs[r] * gv + bv;
        x[(size_t)nrow * DD + n] += y;                     // residual, row owned
      }
    }
  }
}

// ---------------------------------------------------------------------------
// Host orchestration (graph-capture safe: only async ops on `stream`).
// ---------------------------------------------------------------------------
extern "C" void kernel_launch(void* const* d_in, const int* in_sizes, int n_in,
                              void* d_out, int out_size, void* d_ws, size_t ws_size,
                              hipStream_t stream) {
  (void)n_in; (void)out_size; (void)ws_size;

  const float* x_in    = (const float*)d_in[0];
  const float* ef_in   = (const float*)d_in[1];
  const int*   ei      = (const int*)d_in[2];
  const float* eW0     = (const float*)d_in[3];
  const float* eb0     = (const float*)d_in[4];
  const float* eW1     = (const float*)d_in[5];
  const float* eb1     = (const float*)d_in[6];
  const float* eW2     = (const float*)d_in[7];
  const float* eb2     = (const float*)d_in[8];
  const float* eg      = (const float*)d_in[9];
  const float* ebeta   = (const float*)d_in[10];
  const float* nW0     = (const float*)d_in[11];
  const float* nb0     = (const float*)d_in[12];
  const float* nW1     = (const float*)d_in[13];
  const float* nb1     = (const float*)d_in[14];
  const float* nW2     = (const float*)d_in[15];
  const float* nb2     = (const float*)d_in[16];
  const float* ng      = (const float*)d_in[17];
  const float* nbeta   = (const float*)d_in[18];

  const long long N = in_sizes[0] / DD;
  const long long E = in_sizes[2] / 2;

  // Workspace layout
  char* ws = (char*)d_ws;
  size_t off = 0;
  auto wsalloc = [&](size_t bytes) -> void* {
    void* p = ws + off;
    off = (off + bytes + 255) & ~(size_t)255;
    return p;
  };
  float*  x_cur  = (float*)wsalloc((size_t)N * DD * sizeof(float));
  float*  ef_cur = (float*)wsalloc((size_t)E * DD * sizeof(float));
  float*  agg    = (float*)wsalloc((size_t)N * DD * sizeof(float));
  bf16_t* pe0 = (bf16_t*)wsalloc((size_t)NLAYERS * 384 * DD * sizeof(bf16_t));
  bf16_t* pe1 = (bf16_t*)wsalloc((size_t)NLAYERS * 128 * DD * sizeof(bf16_t));
  bf16_t* pe2 = (bf16_t*)wsalloc((size_t)NLAYERS * 128 * DD * sizeof(bf16_t));
  bf16_t* pn0 = (bf16_t*)wsalloc((size_t)NLAYERS * 256 * DD * sizeof(bf16_t));
  bf16_t* pn1 = (bf16_t*)wsalloc((size_t)NLAYERS * 128 * DD * sizeof(bf16_t));
  bf16_t* pn2 = (bf16_t*)wsalloc((size_t)NLAYERS * 128 * DD * sizeof(bf16_t));

  hipMemcpyAsync(x_cur,  x_in,  (size_t)N * DD * sizeof(float),
                 hipMemcpyDeviceToDevice, stream);
  hipMemcpyAsync(ef_cur, ef_in, (size_t)E * DD * sizeof(float),
                 hipMemcpyDeviceToDevice, stream);

  auto packBlocks = [](int K) {
    int tiles = NLAYERS * (K / 32) * 8;
    return (tiles + 7) / 8;   // 8 waves per 256-thread block
  };
  pack_w_kernel<<<packBlocks(384), 256, 0, stream>>>(eW0, pe0, 384);
  pack_w_kernel<<<packBlocks(128), 256, 0, stream>>>(eW1, pe1, 128);
  pack_w_kernel<<<packBlocks(128), 256, 0, stream>>>(eW2, pe2, 128);
  pack_w_kernel<<<packBlocks(256), 256, 0, stream>>>(nW0, pn0, 256);
  pack_w_kernel<<<packBlocks(128), 256, 0, stream>>>(nW1, pn1, 128);
  pack_w_kernel<<<packBlocks(128), 256, 0, stream>>>(nW2, pn2, 128);

  constexpr size_t EDGE_SMEM = (size_t)BM * (384 + 8 + 128 + 8) * sizeof(bf16_t);
  constexpr size_t NODE_SMEM = (size_t)BM * (256 + 8 + 128 + 8) * sizeof(bf16_t);

  const int edgeBlocks = (int)((E + BM - 1) / BM);
  const int nodeBlocks = (int)((N + BM - 1) / BM);
  const int zeroBlocks = (int)((N * DD + 255) / 256);

  for (int l = 0; l < NLAYERS; ++l) {
    zero_f32_kernel<<<zeroBlocks, 256, 0, stream>>>(agg, N * DD);
    edge_mlp_kernel<<<edgeBlocks, 256, EDGE_SMEM, stream>>>(
        x_cur, ef_cur, agg, ei, pe0, pe1, pe2, eb0, eb1, eb2, eg, ebeta, l, E);
    node_mlp_kernel<<<nodeBlocks, 256, NODE_SMEM, stream>>>(
        x_cur, agg, pn0, pn1, pn2, nb0, nb1, nb2, ng, nbeta, l, N);
  }

  hipMemcpyAsync(d_out, x_cur, (size_t)N * DD * sizeof(float),
                 hipMemcpyDeviceToDevice, stream);
}
